// model_28415503630530
// MI455X (gfx1250) — matrix operand order repaired and checked by reading, not compiled, not run on hardware
//
#include <hip/hip_runtime.h>
#include <cstdint>

typedef __attribute__((ext_vector_type(16))) _Float16 v16h;
typedef __attribute__((ext_vector_type(8)))  float    v8f;

#define HW 4096
#define LIN(i) (-1.0f + (float)(i) * (2.0f / 63.0f))

// ---------------------------------------------------------------------------
// Generic small conv (3x3 SAME or 1x1), optional relu, optional residual add.
// Channel stride is HW; batch strides passed explicitly (tensors may be
// sub-channel views of larger buffers).
// ---------------------------------------------------------------------------
__global__ void conv_k(const float* __restrict__ in, int Cin, int inBS,
                       const float* __restrict__ w, const float* __restrict__ bias,
                       float* __restrict__ out, int Cout, int outBS,
                       int k3, int relu, const float* __restrict__ res, int resBS)
{
    int idx = blockIdx.x * blockDim.x + threadIdx.x;
    int total = 4 * Cout * HW;
    if (idx >= total) return;
    int p  = idx & (HW - 1);
    int oc = (idx >> 12) % Cout;
    int b  = idx / (Cout << 12);
    int y = p >> 6, x = p & 63;

    float s = bias[oc];
    const float* ib = in + b * inBS;
    if (k3) {
        for (int ic = 0; ic < Cin; ++ic) {
            const float* ip = ib + ic * HW;
            const float* wp = w + (oc * Cin + ic) * 9;
            #pragma unroll
            for (int ky = 0; ky < 3; ++ky) {
                int yy = y + ky - 1;
                if (yy < 0 || yy > 63) continue;
                #pragma unroll
                for (int kx = 0; kx < 3; ++kx) {
                    int xx = x + kx - 1;
                    if (xx < 0 || xx > 63) continue;
                    s += ip[yy * 64 + xx] * wp[ky * 3 + kx];
                }
            }
        }
    } else {
        for (int ic = 0; ic < Cin; ++ic)
            s += ib[ic * HW + p] * w[oc * Cin + ic];
    }
    if (res) s += res[b * resBS + oc * HW + p];
    if (relu) s = fmaxf(s, 0.0f);
    out[b * outBS + oc * HW + p] = s;
}

__global__ void copych_k(const float* __restrict__ in, int inBS,
                         float* __restrict__ out, int outBS, int C)
{
    int idx = blockIdx.x * blockDim.x + threadIdx.x;
    int total = 4 * C * HW;
    if (idx >= total) return;
    int p = idx & (HW - 1);
    int c = (idx >> 12) % C;
    int b = idx / (C << 12);
    out[b * outBS + c * HW + p] = in[b * inBS + c * HW + p];
}

// ---------------------------------------------------------------------------
// Column sums of exp(20*score - 20); score(n,m)=1/(1+dist) from the fixed
// coordinate grid. Max over n of score is exactly 1 (n==m), so the softmax
// max-shift is the constant 20.
// ---------------------------------------------------------------------------
__global__ __launch_bounds__(256) void colsum_k(float* __restrict__ colsum)
{
    __shared__ float red[256];
    int m = blockIdx.x;
    float cxm = LIN(m >> 6), cym = LIN(m & 63);
    float s = 0.0f;
    for (int n = (int)threadIdx.x; n < HW; n += 256) {
        float dx = LIN(n >> 6) - cxm;
        float dy = LIN(n & 63) - cym;
        float dist = sqrtf(dx * dx + dy * dy);
        float sc = 20.0f * __builtin_amdgcn_rcpf(1.0f + dist);
        s += __expf(sc - 20.0f);
    }
    red[threadIdx.x] = s;
    __syncthreads();
    for (int o = 128; o > 0; o >>= 1) {
        if ((int)threadIdx.x < o) red[threadIdx.x] += red[threadIdx.x + o];
        __syncthreads();
    }
    if (threadIdx.x == 0) colsum[m] = red[0];
}

// ---------------------------------------------------------------------------
// Build the transposed, colsum-scaled f16 token matrix:
// flatT[fg][m], fg = b*8 + f, f: 0=gx 1=gy 2..7=feats. Scaling by 1/colsum[m]
// folds the softmax denominator into the B operand of the WMMA matmul.
// ---------------------------------------------------------------------------
__global__ void flatT_k(const float* __restrict__ feats,
                        const float* __restrict__ colsum,
                        _Float16* __restrict__ flatT)
{
    int idx = blockIdx.x * blockDim.x + threadIdx.x;
    if (idx >= 32 * HW) return;
    int m = idx & (HW - 1);
    int fg = idx >> 12;
    int b = fg >> 3, f = fg & 7;
    float v;
    if (f == 0)      v = LIN(m >> 6);
    else if (f == 1) v = LIN(m & 63);
    else             v = feats[(b * 6 + (f - 2)) * HW + m];
    flatT[idx] = (_Float16)(v / colsum[m]);
}

// ---------------------------------------------------------------------------
// Attention matmul: wf[b,n,f] = sum_m exp(20*score(n,m)-20) * flatT[b*8+f][m].
// Grid: 32 blocks x 256 threads (8 waves); wave w owns n-tile blockIdx*8+w.
// m is processed in 4 chunks of 1024; each chunk's 64KB f16 slab is staged
// into LDS with wave-wide global_load_async_to_lds_b128 (ASYNCcnt path).
// Two v_wmma_f32_16x16x32_f16 per K-chunk cover all 4 batches (8 feats each).
// ---------------------------------------------------------------------------
__global__ __launch_bounds__(256) void attn_k(const _Float16* __restrict__ flatT,
                                              float* __restrict__ wf)
{
    __shared__ _Float16 ldsT[32 * 1024];   // 64KB: rows fg=0..31, 1024 m each

    const int tid  = threadIdx.x;
    const int lane = tid & 31;
    const int wave = tid >> 5;
    const int ntile = blockIdx.x * 8 + wave;
    const int hi  = lane >> 4;
    const int l16 = lane & 15;

    const int   nrow = ntile * 16 + l16;          // A-matrix row for this lane
    const float cxn = LIN(nrow >> 6);
    const float cyn = LIN(nrow & 63);

    v8f acc01 = {};   // batches 0,1
    v8f acc23 = {};   // batches 2,3

    for (int mb = 0; mb < 4; ++mb) {
        __syncthreads();   // previous chunk's LDS reads complete
        // Cooperative async stage: 256 lanes * 16 units * 16B = 64KB.
        #pragma unroll 4
        for (int u = 0; u < 16; ++u) {
            int unit = tid * 16 + u;               // 0..4095 units of 8 halves
            int fg = unit >> 7;                    // 0..31
            int mo = (unit & 127) << 3;            // 0..1016 halves
            unsigned ldsAddr = (unsigned)(uintptr_t)&ldsT[fg * 1024 + mo];
            unsigned goff = (unsigned)((fg * HW + mb * 1024 + mo) * 2);
            asm volatile("global_load_async_to_lds_b128 %0, %1, %2"
                         :: "v"(ldsAddr), "v"(goff), "s"(flatT) : "memory");
        }
        asm volatile("s_wait_asynccnt 0" ::: "memory");
        __syncthreads();

        for (int kc = 0; kc < 32; ++kc) {
            const int mrel = kc * 32;
            const int mabs = mb * 1024 + mrel;
            // A tile: regenerate softmax numerator in-register (f16).
            v16h a;
            #pragma unroll
            for (int h = 0; h < 16; ++h) {
                int K = hi * 16 + h;
                int m = mabs + K;
                float dx = LIN(m >> 6) - cxn;
                float dy = LIN(m & 63) - cyn;
                float dist = sqrtf(dx * dx + dy * dy);
                float e = __expf(20.0f * __builtin_amdgcn_rcpf(1.0f + dist) - 20.0f);
                a[h] = (_Float16)e;
            }
            // B tiles from LDS: column = fg, 16 contiguous K halves per lane.
            const v16h b0 = *(const v16h*)&ldsT[(l16)      * 1024 + mrel + hi * 16];
            const v16h b1 = *(const v16h*)&ldsT[(16 + l16) * 1024 + mrel + hi * 16];
            acc01 = __builtin_amdgcn_wmma_f32_16x16x32_f16(
                        false, a, false, b0, (short)0, acc01, false, false);
            acc23 = __builtin_amdgcn_wmma_f32_16x16x32_f16(
                        false, a, false, b1, (short)0, acc23, false, false);
        }
    }

    // C/D layout: VGPR r -> row r + 8*hi; col = l16 -> (batch, feat).
    const int batch = l16 >> 3;
    const int f = l16 & 7;
    #pragma unroll
    for (int r = 0; r < 8; ++r) {
        int row = ntile * 16 + r + hi * 8;
        wf[((batch)     * HW + row) * 8 + f] = acc01[r];
        wf[((batch + 2) * HW + row) * 8 + f] = acc23[r];
    }
}

// ---------------------------------------------------------------------------
// Token MLP on WMMA: 4 layers of (16x64 @ 64x64) per 16-token tile with
// LayerNorm+ReLU between layers (per-row via LDS f32 scratch), final 64->3.
// Block: 128 threads = 4 waves, each wave owns a 16-token tile.
// ---------------------------------------------------------------------------
__global__ __launch_bounds__(128) void mlp_k(const float* __restrict__ wf,
    const float* __restrict__ w0,  const float* __restrict__ b0v,
    const float* __restrict__ g0,  const float* __restrict__ be0,
    const float* __restrict__ wh,  const float* __restrict__ bh,
    const float* __restrict__ gh,  const float* __restrict__ beh,
    const float* __restrict__ wo,  const float* __restrict__ bo,
    float* __restrict__ out)
{
    __shared__ _Float16 wL[4][64][64];        // layer weights, f16 (layer0 K>=8 zero)
    __shared__ float bias[4][64], gam[4][64], bet[4][64];
    __shared__ float woS[3][64];
    __shared__ float boS[4];
    __shared__ float scratch[4][16][68];      // per-wave f32 pre-LN tile (padded)
    __shared__ _Float16 act[4][16][72];       // per-wave f16 activations (padded)

    const int tid  = threadIdx.x;
    const int lane = tid & 31;
    const int wave = tid >> 5;
    const int hi  = lane >> 4;
    const int l16 = lane & 15;

    // Stage weights/params into LDS (f16 weights for WMMA B operands).
    for (int i = tid; i < 4 * 64 * 64; i += 128) {
        int l = i >> 12, n = (i >> 6) & 63, k = i & 63;
        float v = (l == 0) ? ((k < 8) ? w0[n * 8 + k] : 0.0f)
                           : wh[((l - 1) * 64 + n) * 64 + k];
        wL[l][n][k] = (_Float16)v;
    }
    for (int i = tid; i < 256; i += 128) {
        int l = i >> 6, j = i & 63;
        bias[l][j] = (l == 0) ? b0v[j] : bh[(l - 1) * 64 + j];
        gam[l][j]  = (l == 0) ? g0[j]  : gh[(l - 1) * 64 + j];
        bet[l][j]  = (l == 0) ? be0[j] : beh[(l - 1) * 64 + j];
    }
    if (tid < 192) woS[tid / 64][tid % 64] = wo[tid];
    if (tid < 3)   boS[tid] = bo[tid];

    const int tok0 = (blockIdx.x * 4 + wave) * 16;
    // Init activations from wf (cols >= 8 zero to match padded layer0 weights).
    for (int i = lane; i < 16 * 64; i += 32) {
        int t = i >> 6, k = i & 63;
        act[wave][t][k] = (k < 8) ? (_Float16)wf[(tok0 + t) * 8 + k] : (_Float16)0.0f;
    }
    __syncthreads();

    for (int l = 0; l < 4; ++l) {
        v8f acc[4] = {};
        #pragma unroll
        for (int kc = 0; kc < 2; ++kc) {
            // A: halves 0-7 -> K = kc*32 + hi*8 + h ; halves 8-15 -> +16.
            union { uint4 u[2]; v16h v; } ua;
            ua.u[0] = *(const uint4*)&act[wave][l16][kc * 32 + hi * 8];
            ua.u[1] = *(const uint4*)&act[wave][l16][kc * 32 + 16 + hi * 8];
            #pragma unroll
            for (int t = 0; t < 4; ++t) {
                int ncol = t * 16 + l16;
                union { uint4 u[2]; v16h v; } ub;
                ub.u[0] = *(const uint4*)&wL[l][ncol][kc * 32 + hi * 8];
                ub.u[1] = *(const uint4*)&wL[l][ncol][kc * 32 + 16 + hi * 8];
                acc[t] = __builtin_amdgcn_wmma_f32_16x16x32_f16(
                             false, ua.v, false, ub.v, (short)0, acc[t], false, false);
            }
        }
        // Spill + bias add (C layout: VGPR r -> row r + 8*hi, col = t*16+l16).
        #pragma unroll
        for (int t = 0; t < 4; ++t)
            #pragma unroll
            for (int r = 0; r < 8; ++r) {
                int row = r + hi * 8, col = t * 16 + l16;
                scratch[wave][row][col] = acc[t][r] + bias[l][col];
            }
        __syncthreads();
        // LayerNorm + ReLU, one token row per low lane.
        if (lane < 16) {
            float mu = 0.0f;
            for (int k = 0; k < 64; ++k) mu += scratch[wave][lane][k];
            mu *= (1.0f / 64.0f);
            float var = 0.0f;
            for (int k = 0; k < 64; ++k) {
                float d = scratch[wave][lane][k] - mu;
                var += d * d;
            }
            var *= (1.0f / 64.0f);
            float rs = __builtin_amdgcn_rsqf(var + 1e-5f);
            for (int k = 0; k < 64; ++k) {
                float y = (scratch[wave][lane][k] - mu) * rs * gam[l][k] + bet[l][k];
                act[wave][lane][k] = (_Float16)fmaxf(y, 0.0f);
            }
        }
        __syncthreads();
    }

    // Final projection 64 -> 3.
    if (lane < 16) {
        int t = lane;
        #pragma unroll
        for (int j = 0; j < 3; ++j) {
            float s = boS[j];
            for (int k = 0; k < 64; ++k) s += (float)act[wave][t][k] * woS[j][k];
            out[(tok0 + t) * 3 + j] = s;
        }
    }
}

// ---------------------------------------------------------------------------
extern "C" void kernel_launch(void* const* d_in, const int* in_sizes, int n_in,
                              void* d_out, int out_size, void* d_ws, size_t ws_size,
                              hipStream_t stream)
{
    (void)in_sizes; (void)n_in; (void)out_size; (void)ws_size;
    const float* img    = (const float*)d_in[0];
    const float* sfe1_w = (const float*)d_in[1];
    const float* sfe1_b = (const float*)d_in[2];
    const float* sfe2_w = (const float*)d_in[3];
    const float* sfe2_b = (const float*)d_in[4];
    const float* rdb_w0 = (const float*)d_in[5];
    const float* rdb_b0 = (const float*)d_in[6];
    const float* rdb_w1 = (const float*)d_in[7];
    const float* rdb_b1 = (const float*)d_in[8];
    const float* rdb_w2 = (const float*)d_in[9];
    const float* rdb_b2 = (const float*)d_in[10];
    const float* lff_w  = (const float*)d_in[11];
    const float* lff_b  = (const float*)d_in[12];
    const float* gff1_w = (const float*)d_in[13];
    const float* gff1_b = (const float*)d_in[14];
    const float* gff2_w = (const float*)d_in[15];
    const float* gff2_b = (const float*)d_in[16];
    const float* fc_w0  = (const float*)d_in[17];
    const float* fc_b0  = (const float*)d_in[18];
    const float* ln_g0  = (const float*)d_in[19];
    const float* ln_b0  = (const float*)d_in[20];
    const float* fc_wh  = (const float*)d_in[21];
    const float* fc_bh  = (const float*)d_in[22];
    const float* ln_gh  = (const float*)d_in[23];
    const float* ln_bh  = (const float*)d_in[24];
    const float* fc_wo  = (const float*)d_in[25];
    const float* fc_bo  = (const float*)d_in[26];

    // Workspace carve-up (~5.5 MB of f32).
    float* ws     = (float*)d_ws;
    float* s1     = ws;                 // 4*6*HW
    float* x      = s1 + 4 * 6 * HW;
    float* h      = x  + 4 * 6 * HW;    // 4*24*HW
    float* cat    = h  + 4 * 24 * HW;   // 4*24*HW
    float* gA     = cat + 4 * 24 * HW;  // 4*6*HW
    float* feats  = gA + 4 * 6 * HW;    // 4*6*HW
    float* colsum = feats + 4 * 6 * HW; // HW
    float* wf     = colsum + HW;        // 4*HW*8
    _Float16* flatT = (_Float16*)(wf + 4 * HW * 8);  // 32*HW halves

    auto cdiv = [](int a, int b) { return (a + b - 1) / b; };
    const int BS6 = 6 * HW, BS24 = 24 * HW, BS3 = 3 * HW;
    const int g6 = cdiv(4 * 6 * HW, 256);

    // Shallow feature extraction.
    conv_k<<<g6, 256, 0, stream>>>(img, 3, BS3, sfe1_w, sfe1_b, s1, 6, BS6, 1, 0, nullptr, 0);
    conv_k<<<g6, 256, 0, stream>>>(s1, 6, BS6, sfe2_w, sfe2_b, x,  6, BS6, 1, 0, nullptr, 0);

    // Residual dense blocks.
    for (int d = 0; d < 4; ++d) {
        copych_k<<<g6, 256, 0, stream>>>(x, BS6, h, BS24, 6);
        conv_k<<<g6, 256, 0, stream>>>(h,  6, BS24, rdb_w0 + d * 6 * 6 * 9,  rdb_b0 + d * 6,
                                       h + 6 * HW, 6, BS24, 1, 1, nullptr, 0);
        conv_k<<<g6, 256, 0, stream>>>(h, 12, BS24, rdb_w1 + d * 6 * 12 * 9, rdb_b1 + d * 6,
                                       h + 12 * HW, 6, BS24, 1, 1, nullptr, 0);
        conv_k<<<g6, 256, 0, stream>>>(h, 18, BS24, rdb_w2 + d * 6 * 18 * 9, rdb_b2 + d * 6,
                                       h + 18 * HW, 6, BS24, 1, 1, nullptr, 0);
        conv_k<<<g6, 256, 0, stream>>>(h, 24, BS24, lff_w + d * 6 * 24, lff_b + d * 6,
                                       x, 6, BS6, 0, 0, x, BS6);
        copych_k<<<g6, 256, 0, stream>>>(x, BS6, cat + d * 6 * HW, BS24, 6);
    }

    // Global feature fusion + residual.
    conv_k<<<g6, 256, 0, stream>>>(cat, 24, BS24, gff1_w, gff1_b, gA, 6, BS6, 0, 0, nullptr, 0);
    conv_k<<<g6, 256, 0, stream>>>(gA, 6, BS6, gff2_w, gff2_b, feats, 6, BS6, 1, 0, s1, BS6);

    // Softmax column sums, scaled f16 token matrix, WMMA attention, WMMA MLP.
    colsum_k<<<HW, 256, 0, stream>>>(colsum);
    flatT_k<<<cdiv(32 * HW, 256), 256, 0, stream>>>(feats, colsum, flatT);
    attn_k<<<32, 256, 0, stream>>>(flatT, wf);
    mlp_k<<<128, 128, 0, stream>>>(wf, fc_w0, fc_b0, ln_g0, ln_b0,
                                   fc_wh, fc_bh, ln_gh, ln_bh, fc_wo, fc_bo,
                                   (float*)d_out);
}
